// TransformerBlock_73332271611918
// MI455X (gfx1250) — matrix operand order repaired and checked by reading, not compiled, not run on hardware
//
#include <hip/hip_runtime.h>
#include <math.h>

// Problem dims (fixed by the reference)
#define M_TOK 4096   // B*T tokens
#define T_SEQ 2048
#define DM    1024
#define NH    16
#define DH    64
#define DFF   4096

typedef __attribute__((ext_vector_type(16))) __bf16 v16bf;
typedef __attribute__((ext_vector_type(8)))  float  v8f;
typedef __attribute__((ext_vector_type(4)))  unsigned int u32x4;
typedef __attribute__((ext_vector_type(8)))  unsigned int u32x8;

#define WMMA_BF16(a, b, c) \
  __builtin_amdgcn_wmma_f32_16x16x32_bf16(false, (a), false, (b), (short)0, (c), false, false)

// ---- helpers -------------------------------------------------------------

__device__ __forceinline__ unsigned short f2bf(float f) {
  union { float f; unsigned u; } v; v.f = f;
  unsigned r = v.u + 0x7FFFu + ((v.u >> 16) & 1u);   // round-to-nearest-even
  return (unsigned short)(r >> 16);
}

// A fragment (16x32, 16-bit): lane<16 holds M=lane, K={0..7,16..23};
// lane>=16 holds M=lane-16, K={8..15,24..31}.  tile points at (row0,k0), ld in elems.
__device__ __forceinline__ v16bf frag_a(const unsigned short* tile, int ld, int lane) {
  const unsigned short* p = tile + (size_t)(lane & 15) * ld + ((lane & 16) ? 8 : 0);
  union { uint4 q[2]; v16bf v; } u;
  u.q[0] = *(const uint4*)p;
  u.q[1] = *(const uint4*)(p + 16);
  return u.v;
}

// B fragment (32x16, 16-bit) loaded from row-major [N][K] (i.e. W transposed):
// lane<16 holds N=lane, K={0..7,16..23}; lane>=16 holds N=lane-16, K={8..15,24..31}.
__device__ __forceinline__ v16bf frag_b(const unsigned short* tile, int ld, int lane) {
  const unsigned short* p = tile + (size_t)(lane & 15) * ld + ((lane & 16) ? 8 : 0);
  union { uint4 q[2]; v16bf v; } u;
  u.q[0] = *(const uint4*)p;
  u.q[1] = *(const uint4*)(p + 16);
  return u.v;
}

__device__ __forceinline__ float gelu_tanh(float x) {
  float t = tanhf(0.7978845608028654f * (x + 0.044715f * x * x * x));
  return 0.5f * x * (1.0f + t);
}

__device__ __forceinline__ float redmax16(float v) {
#pragma unroll
  for (int off = 1; off < 16; off <<= 1) v = fmaxf(v, __shfl_xor(v, off, 32));
  return v;
}
__device__ __forceinline__ float redsum16(float v) {
#pragma unroll
  for (int off = 1; off < 16; off <<= 1) v += __shfl_xor(v, off, 32);
  return v;
}

// ---- Tensor Data Mover: 2D bf16 tile (rows x 32) global -> LDS -----------
// D# group0: count=1, lds_addr, global_addr, type=2.  D# group1: data_size=2B,
// tensor_dim0=ld, tile_dim0=32, tile_dim1=rows, tensor_dim0_stride=ld.
__device__ __forceinline__ void tdm_load_tile(void* ldsp, const void* gp,
                                              unsigned ld_elems, unsigned rows) {
  unsigned long long ga = (unsigned long long)gp;
  unsigned lds_off = (unsigned)(unsigned long long)ldsp;   // LDS aperture low bits
  u32x4 g0;
  g0[0] = 1u;                                              // count=1 (valid), user mode
  g0[1] = lds_off;                                         // lds_addr (bytes)
  g0[2] = (unsigned)ga;                                    // global_addr[31:0]
  g0[3] = (unsigned)((ga >> 32) & 0x01FFFFFFu) | (2u << 30); // addr[56:32] | type=2
  unsigned long long st0 = (unsigned long long)ld_elems;   // dim0 stride (elems)
  unsigned td0 = ld_elems;                                 // tensor_dim0 (elems)
  unsigned td1 = 1u << 20;                                 // tensor_dim1 (large, no OOB)
  u32x8 g1;
  g1[0] = (1u << 16);                                      // wg_mask=0, data_size=1 (2B)
  g1[1] = (td0 & 0xFFFFu) << 16;                           // abar_addr=0 | tdim0 lo16
  g1[2] = ((td0 >> 16) & 0xFFFFu) | ((td1 & 0xFFFFu) << 16);
  g1[3] = ((td1 >> 16) & 0xFFFFu) | (32u << 16);           // tile_dim0 = 32
  g1[4] = (rows & 0xFFFFu);                                // tile_dim1 = rows, tile_dim2=0
  g1[5] = (unsigned)(st0 & 0xFFFFFFFFu);                   // dim0_stride lo32
  g1[6] = (unsigned)((st0 >> 32) & 0xFFFFu);               // dim0_stride hi | stride1 lo=0
  g1[7] = 0u;
  asm volatile("tensor_load_to_lds %0, %1" :: "s"(g0), "s"(g1) : "memory");
}

// ---- LayerNorm (one 1024-wide row per 256-thread block) -> bf16 ----------

__global__ __launch_bounds__(256) void k_ln(const float* __restrict__ x,
                                            const float* __restrict__ g,
                                            const float* __restrict__ b,
                                            unsigned short* __restrict__ out) {
  const int row = blockIdx.x;
  const float* xr = x + (size_t)row * DM;
  float s = 0.f, ss = 0.f;
#pragma unroll
  for (int i = 0; i < DM / 256; ++i) {
    float v = xr[threadIdx.x + i * 256]; s += v; ss += v * v;
  }
  __shared__ float shs[8], shss[8];
#pragma unroll
  for (int off = 16; off; off >>= 1) { s += __shfl_xor(s, off, 32); ss += __shfl_xor(ss, off, 32); }
  int wid = threadIdx.x >> 5, lane = threadIdx.x & 31;
  if (lane == 0) { shs[wid] = s; shss[wid] = ss; }
  __syncthreads();
  if (threadIdx.x == 0) {
    float a = 0.f, c = 0.f;
#pragma unroll
    for (int i = 0; i < 8; ++i) { a += shs[i]; c += shss[i]; }
    shs[0] = a; shss[0] = c;
  }
  __syncthreads();
  float mu  = shs[0] * (1.0f / DM);
  float var = shss[0] * (1.0f / DM) - mu * mu;
  float rs  = rsqrtf(var + 1e-5f);
#pragma unroll
  for (int i = 0; i < DM / 256; ++i) {
    int c = threadIdx.x + i * 256;
    float v = (xr[c] - mu) * rs * g[c] + b[c];
    out[(size_t)row * DM + c] = f2bf(v);
  }
}

// ---- weight fp32 [K][N] -> bf16 transposed [N][K] ------------------------

__global__ __launch_bounds__(256) void k_wt(const float* __restrict__ W,
                                            unsigned short* __restrict__ Wt,
                                            int K, int N) {
  long long i = (long long)blockIdx.x * 256 + threadIdx.x;
  if (i >= (long long)K * N) return;
  int k = (int)(i / N), n = (int)(i % N);
  Wt[(size_t)n * K + k] = f2bf(W[i]);
}

// ---- transpose V part of qkv: [M][3DM] cols 2DM.. -> Vt[DM][M] -----------

__global__ __launch_bounds__(256) void k_vt(const unsigned short* __restrict__ qkvb,
                                            unsigned short* __restrict__ Vt) {
  long long i = (long long)blockIdx.x * 256 + threadIdx.x;
  if (i >= (long long)M_TOK * DM) return;
  int m = (int)(i >> 10), c = (int)(i & (DM - 1));
  Vt[(size_t)c * M_TOK + m] = qkvb[(size_t)m * (3 * DM) + 2 * DM + c];
}

// ---- bf16 WMMA GEMM, TDM double-buffered through LDS ---------------------
// out[M][N] = A[M][K] @ Wt[N][K]^T + bias (+epilogue).
// block = 256 thr (8 waves), block tile 64x128, wave tile 32x32.
// Wave 0 DMAs the next A(64x32)/B(128x32) tiles with tensor_load_to_lds while
// all waves run WMMAs on the current LDS buffer (ds_load_b128 fragments).

template <int DO_GELU, int HAS_RES, int OUT_F32>
__global__ __launch_bounds__(256) void k_gemm(const unsigned short* __restrict__ A,
                                              const unsigned short* __restrict__ Wt,
                                              const float* __restrict__ bias,
                                              const float* __restrict__ resid,
                                              float* __restrict__ outF,
                                              unsigned short* __restrict__ outB,
                                              int Nn, int Kk) {
  __shared__ __align__(16) unsigned short ldsA[2][64 * 32];
  __shared__ __align__(16) unsigned short ldsB[2][128 * 32];

  const int lane = threadIdx.x & 31;
  const int wid  = threadIdx.x >> 5;
  const int mblk = blockIdx.y * 64;
  const int nblk = blockIdx.x * 128;
  const int wm   = (wid >> 2) * 32;     // wave row offset inside block tile
  const int wn   = (wid & 3) * 32;      // wave col offset inside block tile

  const unsigned short* Ab = A  + (size_t)mblk * Kk;
  const unsigned short* Bb = Wt + (size_t)nblk * Kk;

  v8f c00 = {}, c01 = {}, c10 = {}, c11 = {};

  if (wid == 0) {
    tdm_load_tile(&ldsA[0][0], Ab, (unsigned)Kk, 64);
    tdm_load_tile(&ldsB[0][0], Bb, (unsigned)Kk, 128);
    __builtin_amdgcn_s_wait_tensorcnt(0);
  }
  __syncthreads();

  int cb = 0;
  for (int k0 = 0; k0 < Kk; k0 += 32) {
    if (k0 + 32 < Kk && wid == 0) {      // prefetch next tiles via TDM
      tdm_load_tile(&ldsA[cb ^ 1][0], Ab + k0 + 32, (unsigned)Kk, 64);
      tdm_load_tile(&ldsB[cb ^ 1][0], Bb + k0 + 32, (unsigned)Kk, 128);
    }
    v16bf a0 = frag_a(&ldsA[cb][(size_t)wm * 32],        32, lane);
    v16bf a1 = frag_a(&ldsA[cb][(size_t)(wm + 16) * 32], 32, lane);
    v16bf b0 = frag_b(&ldsB[cb][(size_t)wn * 32],        32, lane);
    v16bf b1 = frag_b(&ldsB[cb][(size_t)(wn + 16) * 32], 32, lane);
    c00 = WMMA_BF16(a0, b0, c00);
    c01 = WMMA_BF16(a0, b1, c01);
    c10 = WMMA_BF16(a1, b0, c10);
    c11 = WMMA_BF16(a1, b1, c11);
    __syncthreads();                      // everyone done reading buf[cb]
    if (wid == 0) __builtin_amdgcn_s_wait_tensorcnt(0);
    __syncthreads();                      // next buffer published
    cb ^= 1;
  }

  const int colr   = lane & 15;
  const int rowoff = (lane & 16) ? 8 : 0;
  const int m0 = mblk + wm;
  const int n0 = nblk + wn;
  const float bia0 = bias[n0 + colr];
  const float bia1 = bias[n0 + 16 + colr];
#pragma unroll
  for (int r = 0; r < 8; ++r) {
    int rowA = m0 + r + rowoff;
    int rowB = rowA + 16;
    size_t iA0 = (size_t)rowA * Nn + n0 + colr;
    size_t iB0 = (size_t)rowB * Nn + n0 + colr;
    float v0 = c00[r] + bia0, v1 = c01[r] + bia1;
    float v2 = c10[r] + bia0, v3 = c11[r] + bia1;
    if (DO_GELU) { v0 = gelu_tanh(v0); v1 = gelu_tanh(v1); v2 = gelu_tanh(v2); v3 = gelu_tanh(v3); }
    if (HAS_RES) { v0 += resid[iA0]; v1 += resid[iA0 + 16]; v2 += resid[iB0]; v3 += resid[iB0 + 16]; }
    if (OUT_F32) {
      outF[iA0] = v0; outF[iA0 + 16] = v1; outF[iB0] = v2; outF[iB0 + 16] = v3;
    } else {
      outB[iA0] = f2bf(v0); outB[iA0 + 16] = f2bf(v1);
      outB[iB0] = f2bf(v2); outB[iB0 + 16] = f2bf(v3);
    }
  }
}

// ---- flash attention: 1 wave = one (b,h, 16-row q-tile); online softmax ---

__global__ __launch_bounds__(128) void k_attn(const unsigned short* __restrict__ qkvb,
                                              const unsigned short* __restrict__ Vt,
                                              unsigned short* __restrict__ attnb) {
  __shared__ __align__(16) unsigned short sP[4][16 * 32];  // per-wave P staging
  const int lane = threadIdx.x & 31;
  const int widl = threadIdx.x >> 5;
  const int wgid = blockIdx.x * 4 + widl;          // 0..4095
  const int qtile = wgid & 127;
  const int bh    = wgid >> 7;
  const int b = bh >> 4, h = bh & 15;
  const int q0 = qtile << 4;

  const unsigned short* Qb = qkvb + (size_t)(b * T_SEQ + q0) * (3 * DM) + h * DH;
  const unsigned short* Kb = qkvb + (size_t)(b * T_SEQ) * (3 * DM) + DM + h * DH;

  const v16bf aq0 = frag_a(Qb,      3 * DM, lane);  // Q dh 0..31
  const v16bf aq1 = frag_a(Qb + 32, 3 * DM, lane);  // Q dh 32..63

  v8f o0 = {}, o1 = {}, o2 = {}, o3 = {};
  v8f m_i, l_i;
#pragma unroll
  for (int r = 0; r < 8; ++r) { m_i[r] = -1e30f; l_i[r] = 0.f; }

  const int colr   = lane & 15;
  const int rowoff = (lane & 16) ? 8 : 0;
  unsigned short* myP = sP[widl];

  for (int j0 = 0; j0 <= q0 + 15; j0 += 32) {
    // S tiles for keys [j0, j0+16) and [j0+16, j0+32): 4 WMMAs over dh=64
    v8f s0 = {}, s1 = {};
    {
      v16bf bk0  = frag_b(Kb + (size_t)j0 * (3 * DM),             3 * DM, lane);
      v16bf bk0b = frag_b(Kb + (size_t)j0 * (3 * DM) + 32,        3 * DM, lane);
      v16bf bk1  = frag_b(Kb + (size_t)(j0 + 16) * (3 * DM),      3 * DM, lane);
      v16bf bk1b = frag_b(Kb + (size_t)(j0 + 16) * (3 * DM) + 32, 3 * DM, lane);
      s0 = WMMA_BF16(aq0, bk0, s0);  s0 = WMMA_BF16(aq1, bk0b, s0);
      s1 = WMMA_BF16(aq0, bk1, s1);  s1 = WMMA_BF16(aq1, bk1b, s1);
    }
    const int jt0 = j0 + colr, jt1 = jt0 + 16;
#pragma unroll
    for (int r = 0; r < 8; ++r) {
      int qt = q0 + r + rowoff;
      float v0 = 0.125f * s0[r]; if (jt0 > qt) v0 = -1e30f;   // 1/sqrt(64) + causal
      float v1 = 0.125f * s1[r]; if (jt1 > qt) v1 = -1e30f;
      float rm = redmax16(fmaxf(v0, v1));
      float mnew = fmaxf(m_i[r], rm);
      float corr = __expf(m_i[r] - mnew);
      float p0 = __expf(v0 - mnew);
      float p1 = __expf(v1 - mnew);
      float rs = redsum16(p0 + p1);
      l_i[r] = l_i[r] * corr + rs;
      m_i[r] = mnew;
      o0[r] *= corr; o1[r] *= corr; o2[r] *= corr; o3[r] *= corr;
      int mrow = r + rowoff;                           // C-layout -> LDS 16x32 tile
      myP[mrow * 32 + colr]      = f2bf(p0);
      myP[mrow * 32 + 16 + colr] = f2bf(p1);
    }
    // Re-read P in A-fragment layout (DS ops are in-order within a wave)
    v16bf ap = frag_a(myP, 32, lane);
    size_t kcol = (size_t)(b * T_SEQ + j0);
    v16bf bv0 = frag_b(Vt + (size_t)(h * DH +  0) * M_TOK + kcol, M_TOK, lane);
    v16bf bv1 = frag_b(Vt + (size_t)(h * DH + 16) * M_TOK + kcol, M_TOK, lane);
    v16bf bv2 = frag_b(Vt + (size_t)(h * DH + 32) * M_TOK + kcol, M_TOK, lane);
    v16bf bv3 = frag_b(Vt + (size_t)(h * DH + 48) * M_TOK + kcol, M_TOK, lane);
    o0 = WMMA_BF16(ap, bv0, o0);
    o1 = WMMA_BF16(ap, bv1, o1);
    o2 = WMMA_BF16(ap, bv2, o2);
    o3 = WMMA_BF16(ap, bv3, o3);
  }

#pragma unroll
  for (int r = 0; r < 8; ++r) {
    int row = b * T_SEQ + q0 + r + rowoff;
    float inv = 1.0f / l_i[r];
    size_t base = (size_t)row * DM + h * DH + colr;
    attnb[base]      = f2bf(o0[r] * inv);
    attnb[base + 16] = f2bf(o1[r] * inv);
    attnb[base + 32] = f2bf(o2[r] * inv);
    attnb[base + 48] = f2bf(o3[r] * inv);
  }
}

// ---- launcher ------------------------------------------------------------

extern "C" void kernel_launch(void* const* d_in, const int* in_sizes, int n_in,
                              void* d_out, int out_size, void* d_ws, size_t ws_size,
                              hipStream_t stream) {
  (void)in_sizes; (void)n_in; (void)out_size; (void)ws_size;
  const float* x      = (const float*)d_in[0];
  const float* qkv_w  = (const float*)d_in[1];
  const float* qkv_b  = (const float*)d_in[2];
  const float* proj_w = (const float*)d_in[3];
  const float* proj_b = (const float*)d_in[4];
  const float* ln1_g  = (const float*)d_in[5];
  const float* ln1_b  = (const float*)d_in[6];
  const float* ln2_g  = (const float*)d_in[7];
  const float* ln2_b  = (const float*)d_in[8];
  const float* fc1_w  = (const float*)d_in[9];
  const float* fc1_b  = (const float*)d_in[10];
  const float* fc2_w  = (const float*)d_in[11];
  const float* fc2_b  = (const float*)d_in[12];
  float* out = (float*)d_out;

  char* w = (char*)d_ws;
  auto aus = [&](size_t n) { unsigned short* p = (unsigned short*)w; w += n * sizeof(unsigned short); return p; };
  auto af  = [&](size_t n) { float* p = (float*)w; w += n * sizeof(float); return p; };

  unsigned short* qkvWt  = aus((size_t)3 * DM * DM);   // [3DM][DM]
  unsigned short* projWt = aus((size_t)DM * DM);       // [DM][DM]
  unsigned short* fc1Wt  = aus((size_t)DFF * DM);      // [DFF][DM]
  unsigned short* fc2Wt  = aus((size_t)DM * DFF);      // [DM][DFF]
  unsigned short* xn1    = aus((size_t)M_TOK * DM);
  unsigned short* qkvb   = aus((size_t)M_TOK * 3 * DM);
  unsigned short* Vtb    = aus((size_t)DM * M_TOK);
  unsigned short* attnb  = aus((size_t)M_TOK * DM);
  unsigned short* xn2    = aus((size_t)M_TOK * DM);
  unsigned short* h1     = aus((size_t)M_TOK * DFF);
  float* xres            = af((size_t)M_TOK * DM);

  // weights -> bf16, transposed
  { long long n = (long long)DM * 3 * DM; k_wt<<<(unsigned)((n + 255) / 256), 256, 0, stream>>>(qkv_w, qkvWt, DM, 3 * DM); }
  { long long n = (long long)DM * DM;     k_wt<<<(unsigned)((n + 255) / 256), 256, 0, stream>>>(proj_w, projWt, DM, DM); }
  { long long n = (long long)DM * DFF;    k_wt<<<(unsigned)((n + 255) / 256), 256, 0, stream>>>(fc1_w, fc1Wt, DM, DFF); }
  { long long n = (long long)DFF * DM;    k_wt<<<(unsigned)((n + 255) / 256), 256, 0, stream>>>(fc2_w, fc2Wt, DFF, DM); }

  // ln1 -> bf16
  k_ln<<<M_TOK, 256, 0, stream>>>(x, ln1_g, ln1_b, xn1);

  // qkv = xn1 @ qkv_w + b   (bf16 out)
  k_gemm<0, 0, 0><<<dim3(3 * DM / 128, M_TOK / 64), 256, 0, stream>>>(
      xn1, qkvWt, qkv_b, nullptr, nullptr, qkvb, 3 * DM, DM);

  // V -> transposed [DM][M]
  { long long n = (long long)M_TOK * DM; k_vt<<<(unsigned)((n + 255) / 256), 256, 0, stream>>>(qkvb, Vtb); }

  // flash attention (bf16 out)
  k_attn<<<(32 * 128) / 4, 128, 0, stream>>>(qkvb, Vtb, attnb);

  // x_res = x + attn @ proj_w + b  (f32 out)
  k_gemm<0, 1, 1><<<dim3(DM / 128, M_TOK / 64), 256, 0, stream>>>(
      attnb, projWt, proj_b, x, xres, nullptr, DM, DM);

  // ln2 -> bf16
  k_ln<<<M_TOK, 256, 0, stream>>>(xres, ln2_g, ln2_b, xn2);

  // h1 = gelu(xn2 @ fc1_w + b)  (bf16 out)
  k_gemm<1, 0, 0><<<dim3(DFF / 128, M_TOK / 64), 256, 0, stream>>>(
      xn2, fc1Wt, fc1_b, nullptr, nullptr, h1, DFF, DM);

  // out = x_res + h1 @ fc2_w + b  (f32 out)
  k_gemm<0, 1, 1><<<dim3(DM / 128, M_TOK / 64), 256, 0, stream>>>(
      h1, fc2Wt, fc2_b, xres, out, nullptr, DM, DFF);
}
